// GaussiansGenerator_79809082295196
// MI455X (gfx1250) — compile-verified, matrix operand order checked by reading, not run on hardware
//
#include <hip/hip_runtime.h>
#include <math.h>

// ---------------- problem constants ----------------
#define BB    4
#define NPTS  4096
#define BNC   (BB*NPTS)        // 16384 point-columns
#define KNB   16               // k neighbors
#define NKBN  (KNB*BNC)        // 262144 cols for [B,C,N,K] tensors
#define DIMC  128
#define NZC   128
#define CHC   640

// ---------------- WMMA / TDM types ----------------
typedef __attribute__((ext_vector_type(16))) __bf16 v16bf_t;
typedef __attribute__((ext_vector_type(8)))  float  v8f_t;
typedef __attribute__((ext_vector_type(4)))  unsigned int u32x4_t;
typedef __attribute__((ext_vector_type(4)))  int    i32x4_t;
typedef __attribute__((ext_vector_type(8)))  int    i32x8_t;

#ifndef USE_TDM
#if defined(__has_builtin)
#if __has_builtin(__builtin_amdgcn_tensor_load_to_lds) && __has_builtin(__builtin_amdgcn_s_wait_tensorcnt)
#define USE_TDM 1
#endif
#endif
#endif
#ifndef USE_TDM
#define USE_TDM 0
#endif

__device__ __forceinline__ unsigned short f2bf(float f) {
  union { float f; unsigned u; } v; v.f = f;
  unsigned r = v.u + 0x7FFFu + ((v.u >> 16) & 1u);   // round-to-nearest-even
  return (unsigned short)(r >> 16);
}

template<int ACT> __device__ __forceinline__ float actf(float x) {
  if (ACT == 1) return x >= 0.f ? x : 0.01f * x;     // lrelu NEG
  if (ACT == 2) return x >= 0.f ? x : 0.2f  * x;     // lrelu NEG2
  if (ACT == 3) return x / (1.f + expf(-x));         // SiLU
  return x;
}

#if USE_TDM
// 1-D TDM copy: nelem f32 from global -> LDS (descriptor per CDNA5 ISA 8.3/8.4)
__device__ __forceinline__ void tdm_load_1d(unsigned lds_off, const void* gptr, unsigned nelem)
{
  unsigned long long ga = (unsigned long long)gptr;
  u32x4_t g0;
  g0[0] = 1u;                                              // count=1, user mode
  g0[1] = lds_off;                                         // lds_addr (bytes)
  g0[2] = (unsigned)(ga & 0xFFFFFFFFu);                    // global_addr[31:0]
  g0[3] = (unsigned)((ga >> 32) & 0x01FFFFFFu) | 0x80000000u; // addr[56:32] | type=2
  i32x8_t g1;
  g1[0] = (int)(2u << 16);                                 // wg_mask=0, data_size=4B
  g1[1] = (int)((nelem & 0xFFFFu) << 16);                  // tensor_dim0[15:0]
  g1[2] = (int)((nelem >> 16) & 0xFFFFu);                  // tensor_dim0[31:16]
  g1[3] = (int)((nelem & 0xFFFFu) << 16);                  // tile_dim0
  g1[4] = 0; g1[5] = 0; g1[6] = 0; g1[7] = 0;
  i32x4_t gz = {0, 0, 0, 0};
#if defined(__clang_major__) && (__clang_major__ >= 23)
  i32x8_t gz8 = {0, 0, 0, 0, 0, 0, 0, 0};
  __builtin_amdgcn_tensor_load_to_lds(g0, g1, gz, gz, gz8, 0);
#else
  __builtin_amdgcn_tensor_load_to_lds(g0, g1, gz, gz, 0);
#endif
}
#endif

// ======================================================================
// Generic bf16 WMMA GEMM:  C[M x Ncol] = act(A * B + bias)
//  OPA=0: A row-major [M][lda] (A[m*lda+k]); OPA=1: A is [K][lda] (A[k*lda+m])
//  OPB=0: B is [K][ldb] (B[k*ldb+col]);      OPB=1: B is [Ncol][ldb] (B[col*ldb+k])
//  REQUIRES Ncol % 128 == 0.  RG=0 additionally requires M%64==0 && K%32==0
//  (guard-free staging + epilogue); RG=1 keeps per-element guards.
//  Block: 256 thr (8 waves), tile 64(M) x 128(N), K-step 32;
//  each wave: 1 A fragment x 4 B fragments -> 4 v_wmma per chunk.
// ======================================================================
#define LDSTR 40   // shorts per LDS row (80B, multiple of 16B)

template<int OPA, int OPB, int ACT, int RG>
__global__ __launch_bounds__(256)
void gemm_bf16(const float* __restrict__ A, const float* __restrict__ B,
               const float* __restrict__ bias, float* __restrict__ C,
               int M, int Ncol, int Kd,
               long lda, long ldb, long ldc,
               long sA, long sB, long sC)
{
  __shared__ unsigned short lA[64  * LDSTR];
  __shared__ unsigned short lB[128 * LDSTR];

  const int tid  = threadIdx.x;
  const int lane = tid & 31;
  const int wave = tid >> 5;
  const int s    = lane >> 4;      // half-wave select
  const int rl   = lane & 15;

  const float* Ap = A + (size_t)blockIdx.z * sA;
  const float* Bp = B + (size_t)blockIdx.z * sB;
  float*       Cp = C + (size_t)blockIdx.z * sC;

  const int m0 = blockIdx.y * 64;
  const int n0 = blockIdx.x * 128;

  const int wr  = wave & 3;            // wave's 16-row tile within 64
  const int wcb = (wave >> 2) * 64;    // wave's 64-col half; 4 tiles of 16

  v8f_t acc0 = {}, acc1 = {}, acc2 = {}, acc3 = {};

  const bool fullM = (m0 + 64 <= M);

  for (int k0 = 0; k0 < Kd; k0 += 32) {
    const bool fast = RG ? (fullM && (k0 + 32 <= Kd)) : true;

    // ---------------- stage A tile (64 x 32) ----------------
    if (OPA == 0) {
      int row = tid >> 2, i0 = (tid & 3) * 8;
      const float* src = Ap + (size_t)(m0 + row) * lda + k0 + i0;
      unsigned short t[8];
      if (fast) {
        float v[8];
        #pragma unroll
        for (int i = 0; i < 8; ++i) v[i] = src[i];
        if (k0 + 64 <= Kd) __builtin_prefetch(src + 32, 0, 3);
        #pragma unroll
        for (int i = 0; i < 8; ++i) t[i] = f2bf(v[i]);
      } else {
        int gm = m0 + row;
        #pragma unroll
        for (int i = 0; i < 8; ++i) {
          int kk = k0 + i0 + i;
          t[i] = (gm < M && kk < Kd) ? f2bf(Ap[(size_t)gm * lda + kk]) : (unsigned short)0;
        }
      }
      uint4 u;
      u.x = (unsigned)t[0] | ((unsigned)t[1] << 16);
      u.y = (unsigned)t[2] | ((unsigned)t[3] << 16);
      u.z = (unsigned)t[4] | ((unsigned)t[5] << 16);
      u.w = (unsigned)t[6] | ((unsigned)t[7] << 16);
      *(uint4*)&lA[row * LDSTR + i0] = u;     // 16B aligned
    } else {
      int kk = tid & 31, mg = tid >> 5, kg = k0 + kk;
      const float* src = Ap + (size_t)kg * lda + m0 + mg * 8;
      if (fast) {
        float v[8];
        #pragma unroll
        for (int i = 0; i < 8; ++i) v[i] = src[i];
        if (k0 + 64 <= Kd) __builtin_prefetch(src + 32 * lda, 0, 3);
        #pragma unroll
        for (int i = 0; i < 8; ++i) lA[(mg * 8 + i) * LDSTR + kk] = f2bf(v[i]);
      } else {
        #pragma unroll
        for (int i = 0; i < 8; ++i) {
          int gm = m0 + mg * 8 + i;
          lA[(mg * 8 + i) * LDSTR + kk] =
              (gm < M && kg < Kd) ? f2bf(Ap[(size_t)kg * lda + gm]) : (unsigned short)0;
        }
      }
    }

    // ---------------- stage B tile (32 x 128), LDS as lB[col][k] ----------------
    if (OPB == 0) {
      int kk = tid & 31, cg = tid >> 5, kg = k0 + kk;   // 8 groups x 16 cols
      const float* src = Bp + (size_t)kg * ldb + n0 + cg * 16;
      if (!RG || (k0 + 32 <= Kd)) {
        float v[16];
        #pragma unroll
        for (int i = 0; i < 16; ++i) v[i] = src[i];
        if (k0 + 64 <= Kd) __builtin_prefetch(src + 32 * ldb, 0, 3);
        #pragma unroll
        for (int i = 0; i < 16; ++i) lB[(cg * 16 + i) * LDSTR + kk] = f2bf(v[i]);
      } else {
        #pragma unroll
        for (int i = 0; i < 16; ++i)
          lB[(cg * 16 + i) * LDSTR + kk] =
              (kg < Kd) ? f2bf(Bp[(size_t)kg * ldb + n0 + cg * 16 + i]) : (unsigned short)0;
      }
    } else {
      int col = tid >> 1, i0 = (tid & 1) * 16;          // 16 contiguous k per thread
      const float* src = Bp + (size_t)(n0 + col) * ldb + k0 + i0;
      unsigned short t[16];
      if (!RG || (k0 + 32 <= Kd)) {
        float v[16];
        #pragma unroll
        for (int i = 0; i < 16; ++i) v[i] = src[i];
        if (k0 + 64 <= Kd) __builtin_prefetch(src + 32, 0, 3);
        #pragma unroll
        for (int i = 0; i < 16; ++i) t[i] = f2bf(v[i]);
      } else {
        #pragma unroll
        for (int i = 0; i < 16; ++i) {
          int kk = k0 + i0 + i;
          t[i] = (kk < Kd) ? f2bf(src[i]) : (unsigned short)0;
        }
      }
      uint4 u0, u1;
      u0.x = (unsigned)t[0]  | ((unsigned)t[1]  << 16);
      u0.y = (unsigned)t[2]  | ((unsigned)t[3]  << 16);
      u0.z = (unsigned)t[4]  | ((unsigned)t[5]  << 16);
      u0.w = (unsigned)t[6]  | ((unsigned)t[7]  << 16);
      u1.x = (unsigned)t[8]  | ((unsigned)t[9]  << 16);
      u1.y = (unsigned)t[10] | ((unsigned)t[11] << 16);
      u1.z = (unsigned)t[12] | ((unsigned)t[13] << 16);
      u1.w = (unsigned)t[14] | ((unsigned)t[15] << 16);
      *(uint4*)&lB[col * LDSTR + i0]     = u0;   // 16B aligned
      *(uint4*)&lB[col * LDSTR + i0 + 8] = u1;
    }
    __syncthreads();

    // ---------------- fragments + 4 x WMMA ----------------
    union Frag { v16bf_t v; unsigned short u[16]; };
    Frag af;
    {
      const unsigned short* p = &lA[(wr * 16 + rl) * LDSTR];
      #pragma unroll
      for (int i = 0; i < 8; ++i) af.u[i]     = p[8 * s + i];
      #pragma unroll
      for (int i = 0; i < 8; ++i) af.u[8 + i] = p[16 + 8 * s + i];
    }
    Frag bf[4];
    #pragma unroll
    for (int j = 0; j < 4; ++j) {
      const unsigned short* p = &lB[(wcb + 16 * j + rl) * LDSTR];
      #pragma unroll
      for (int i = 0; i < 8; ++i) bf[j].u[i]     = p[8 * s + i];
      #pragma unroll
      for (int i = 0; i < 8; ++i) bf[j].u[8 + i] = p[16 + 8 * s + i];
    }
    acc0 = __builtin_amdgcn_wmma_f32_16x16x32_bf16(false, af.v, false, bf[0].v,
                                                   (short)0, acc0, false, false);
    acc1 = __builtin_amdgcn_wmma_f32_16x16x32_bf16(false, af.v, false, bf[1].v,
                                                   (short)0, acc1, false, false);
    acc2 = __builtin_amdgcn_wmma_f32_16x16x32_bf16(false, af.v, false, bf[2].v,
                                                   (short)0, acc2, false, false);
    acc3 = __builtin_amdgcn_wmma_f32_16x16x32_bf16(false, af.v, false, bf[3].v,
                                                   (short)0, acc3, false, false);
    __syncthreads();
  }

  // ---- epilogue: C layout (VGPR i -> M = i + 8*s, N = lane&15) ----
  #pragma unroll
  for (int i = 0; i < 8; ++i) {
    int gm = m0 + wr * 16 + i + 8 * s;
    if (RG && gm >= M) continue;
    float bv = bias ? bias[gm] : 0.f;
    size_t rowb = (size_t)gm * ldc;
    int nb = n0 + wcb + rl;
    Cp[rowb + nb]      = actf<ACT>(acc0[i] + bv);
    Cp[rowb + nb + 16] = actf<ACT>(acc1[i] + bv);
    Cp[rowb + nb + 32] = actf<ACT>(acc2[i] + bv);
    Cp[rowb + nb + 48] = actf<ACT>(acc3[i] + bv);
  }
}

template<int OPA, int OPB, int ACT, int RG = 0>
static inline void gemm(hipStream_t st, const float* A, const float* B,
                        const float* bias, float* C,
                        int M, int Ncol, int Kd,
                        long lda, long ldb, long ldc,
                        long sA = 0, long sB = 0, long sC = 0, int batches = 1)
{
  dim3 g((unsigned)(Ncol / 128), (unsigned)((M + 63) / 64), (unsigned)batches);
  gemm_bf16<OPA, OPB, ACT, RG><<<g, dim3(256), 0, st>>>(A, B, bias, C, M, Ncol, Kd,
                                                        lda, ldb, ldc, sA, sB, sC);
}

// ======================================================================
// Auxiliary kernels
// ======================================================================

// build style input S[131][BNC]: rows 0..2 = x, rows 3..130 = z / ||z||
__global__ void k_prep_style(const float* __restrict__ x, const float* __restrict__ z,
                             float* __restrict__ S)
{
  int col = blockIdx.x * 256 + threadIdx.x;
  if (col >= BNC) return;
  const float* zp = z + (size_t)col * NZC;
  float nrm = 0.f;
  for (int i = 0; i < NZC; ++i) nrm += zp[i] * zp[i];
  nrm = sqrtf(nrm) + 1e-8f;
  for (int d = 0; d < 3; ++d) S[(size_t)d * BNC + col] = x[(size_t)col * 3 + d];
  for (int i = 0; i < NZC; ++i) S[(size_t)(3 + i) * BNC + col] = zp[i] / nrm;
}

// pc conv1: [64][BNC] = lrelu(W[64x3] * x + b)
__global__ void k_pc1(const float* __restrict__ x, const float* __restrict__ w,
                      const float* __restrict__ b, float* __restrict__ out)
{
  int col = blockIdx.x * 256 + threadIdx.x;
  if (col >= BNC) return;
  float x0 = x[(size_t)col * 3], x1 = x[(size_t)col * 3 + 1], x2 = x[(size_t)col * 3 + 2];
  for (int c = 0; c < 64; ++c) {
    float v = w[c * 3] * x0 + w[c * 3 + 1] * x1 + w[c * 3 + 2] * x2 + b[c];
    out[(size_t)c * BNC + col] = v >= 0.f ? v : 0.01f * v;
  }
}

// per-channel batch stats over ncols
__global__ void k_stats(const float* __restrict__ x, float* __restrict__ mean,
                        float* __restrict__ rstd, long ncols)
{
  int c = blockIdx.x;
  const float* p = x + (size_t)c * ncols;
  float s = 0.f, q = 0.f;
  for (long i = threadIdx.x; i < ncols; i += 256) { float v = p[i]; s += v; q += v * v; }
  __shared__ float sh[256], sh2[256];
  sh[threadIdx.x] = s; sh2[threadIdx.x] = q; __syncthreads();
  for (int o = 128; o > 0; o >>= 1) {
    if (threadIdx.x < o) { sh[threadIdx.x] += sh[threadIdx.x + o]; sh2[threadIdx.x] += sh2[threadIdx.x + o]; }
    __syncthreads();
  }
  if (threadIdx.x == 0) {
    float m = sh[0] / (float)ncols;
    float var = sh2[0] / (float)ncols - m * m;
    mean[c] = m; rstd[c] = rsqrtf(var + 1e-5f);
  }
}

// y = lrelu(g*(x-m)*rstd + b) in place; grid.y = channel
__global__ void k_bn_act(float* __restrict__ x, const float* __restrict__ mean,
                         const float* __restrict__ rstd, const float* __restrict__ g,
                         const float* __restrict__ b, long ncols, float slope)
{
  int c = blockIdx.y;
  long i = (long)blockIdx.x * 256 + threadIdx.x;
  if (i >= ncols) return;
  float* p = x + (size_t)c * ncols;
  float v = (p[i] - mean[c]) * rstd[c] * g[c] + b[c];
  p[i] = v >= 0.f ? v : slope * v;
}

// s2[col] = sum_c f[c][col]^2
__global__ void k_sumsq(const float* __restrict__ f, float* __restrict__ s2)
{
  int col = blockIdx.x * 256 + threadIdx.x;
  if (col >= BNC) return;
  float s = 0.f;
  for (int c = 0; c < DIMC; ++c) { float v = f[(size_t)c * BNC + col]; s += v * v; }
  s2[col] = s;
}

// 16 nearest neighbors (excluding self) from Gram + norms
__global__ void k_topk(const float* __restrict__ G, const float* __restrict__ s2,
                       int* __restrict__ idx)
{
  int gid = blockIdx.x * 256 + threadIdx.x;
  if (gid >= BB * NPTS) return;
  int b = gid >> 12, n = gid & (NPTS - 1);
  const float* Grow = G + ((size_t)b * NPTS + n) * NPTS;
  const float* sp = s2 + (size_t)b * NPTS;
  float bd[16]; int bi[16];
  #pragma unroll
  for (int j = 0; j < 16; ++j) { bd[j] = 1e30f; bi[j] = 0; }
  for (int m = 0; m < NPTS; ++m) {
    if (m == n) continue;
    float d = sp[m] - 2.f * Grow[m];
    if (d < bd[15]) {
      bd[15] = d; bi[15] = m;
      #pragma unroll
      for (int j = 15; j > 0; --j) {
        if (bd[j] < bd[j - 1]) {
          float td = bd[j]; bd[j] = bd[j - 1]; bd[j - 1] = td;
          int   ti = bi[j]; bi[j] = bi[j - 1]; bi[j - 1] = ti;
        }
      }
    }
  }
  for (int j = 0; j < 16; ++j) idx[(size_t)gid * 16 + j] = bi[j];
}

// ef[256][16][BNC]: rows 0..127 = center, rows 128..255 = nbr - center
__global__ void k_gather(const float* __restrict__ f, const int* __restrict__ idx,
                         float* __restrict__ ef)
{
  long gid = (long)blockIdx.x * 256 + threadIdx.x;
  if (gid >= (long)KNB * BNC) return;
  int kk = (int)(gid / BNC), col = (int)(gid % BNC);
  int b = col / NPTS;
  int j = idx[(size_t)col * 16 + kk];
  int ncol = b * NPTS + j;
  for (int c = 0; c < DIMC; ++c) {
    float ce = f[(size_t)c * BNC + col];
    float nb = f[(size_t)c * BNC + ncol];
    ef[((size_t)c * KNB + kk) * BNC + col] = ce;
    ef[((size_t)(DIMC + c) * KNB + kk) * BNC + col] = nb - ce;
  }
}

// softmax over the k dimension of w[128][16][BNC]
__global__ void k_softmax_k(float* __restrict__ w)
{
  long gid = (long)blockIdx.x * 256 + threadIdx.x;
  if (gid >= (long)DIMC * BNC) return;
  int c = (int)(gid / BNC), col = (int)(gid % BNC);
  float v[KNB];
  size_t base = (size_t)c * KNB * BNC + col;
  float mx = -1e30f;
  #pragma unroll
  for (int k = 0; k < KNB; ++k) { v[k] = w[base + (size_t)k * BNC]; mx = fmaxf(mx, v[k]); }
  float s = 0.f;
  #pragma unroll
  for (int k = 0; k < KNB; ++k) { v[k] = expf(v[k] - mx); s += v[k]; }
  float inv = 1.f / s;
  #pragma unroll
  for (int k = 0; k < KNB; ++k) w[base + (size_t)k * BNC] = v[k] * inv;
}

__global__ void k_mul(float* __restrict__ a, const float* __restrict__ b, long n)
{
  long i = (long)blockIdx.x * 256 + threadIdx.x;
  if (i < n) a[i] *= b[i];
}

// instance-norm over N + AdaIN modulation; one block per (b,c)
__global__ void k_adain(const float* __restrict__ xin, const float* __restrict__ sb,
                        float* __restrict__ out)
{
  int bc = blockIdx.x, b = bc / DIMC, c = bc % DIMC;
  const float* p = xin + (size_t)c * BNC + (size_t)b * NPTS;
  __shared__ float sh[256], sh2[256];
  float s = 0.f, q = 0.f;
  for (int i = threadIdx.x; i < NPTS; i += 256) { float v = p[i]; s += v; q += v * v; }
  sh[threadIdx.x] = s; sh2[threadIdx.x] = q; __syncthreads();
  for (int o = 128; o > 0; o >>= 1) {
    if (threadIdx.x < o) { sh[threadIdx.x] += sh[threadIdx.x + o]; sh2[threadIdx.x] += sh2[threadIdx.x + o]; }
    __syncthreads();
  }
  float m = sh[0] / (float)NPTS;
  float var = sh2[0] / (float)NPTS - m * m;
  float r = rsqrtf(var + 1e-5f);
  const float* s1 = sb + (size_t)c * BNC + (size_t)b * NPTS;
  const float* s2p = sb + (size_t)(DIMC + c) * BNC + (size_t)b * NPTS;
  float* o = out + (size_t)c * BNC + (size_t)b * NPTS;
  for (int i = threadIdx.x; i < NPTS; i += 256)
    o[i] = s1[i] * ((p[i] - m) * r) + s2p[i];
}

// fg[b*128+c] = max_n x2[c][b*N+n]
__global__ void k_rowmax(const float* __restrict__ x2, float* __restrict__ fg)
{
  int bc = blockIdx.x, b = bc / DIMC, c = bc % DIMC;
  const float* p = x2 + (size_t)c * BNC + (size_t)b * NPTS;
  __shared__ float sh[256];
  float m = -1e30f;
  for (int i = threadIdx.x; i < NPTS; i += 256) m = fmaxf(m, p[i]);
  sh[threadIdx.x] = m; __syncthreads();
  for (int o = 128; o > 0; o >>= 1) {
    if (threadIdx.x < o) sh[threadIdx.x] = fmaxf(sh[threadIdx.x], sh[threadIdx.x + o]);
    __syncthreads();
  }
  if (threadIdx.x == 0) fg[b * DIMC + c] = sh[0];
}

// tiny FC: out[b][o] = sum_c in[b][c]*w[o][c] + bias[o]
__global__ void k_fc_small(const float* __restrict__ in, const float* __restrict__ w,
                           const float* __restrict__ bias, float* __restrict__ out,
                           int Bn, int Cin, int Cout)
{
  int t = blockIdx.x * 256 + threadIdx.x;
  if (t >= Bn * Cout) return;
  int b = t / Cout, o = t % Cout;
  float acc = bias[o];
  for (int c = 0; c < Cin; ++c) acc += in[b * Cin + c] * w[o * Cin + c];
  out[t] = acc;
}

// BatchNorm1d over B rows + lrelu, in place; one thread per channel
__global__ void k_bn1d(float* __restrict__ xv, const float* __restrict__ g,
                       const float* __restrict__ b, int Bn, int C, float slope)
{
  int o = blockIdx.x * 256 + threadIdx.x;
  if (o >= C) return;
  float m = 0.f;
  for (int i = 0; i < Bn; ++i) m += xv[i * C + o];
  m /= (float)Bn;
  float v = 0.f;
  for (int i = 0; i < Bn; ++i) { float d = xv[i * C + o] - m; v += d * d; }
  v /= (float)Bn;
  float r = rsqrtf(v + 1e-5f);
  for (int i = 0; i < Bn; ++i) {
    float y = g[o] * (xv[i * C + o] - m) * r + b[o];
    xv[i * C + o] = y >= 0.f ? y : slope * y;
  }
}

// feat[640][BNC]: rows 0..511 broadcast fg2, rows 512..639 = x2
__global__ void k_feat_build(const float* __restrict__ fg2, const float* __restrict__ x2,
                             float* __restrict__ feat)
{
  long gid = (long)blockIdx.x * 256 + threadIdx.x;
  if (gid >= (long)CHC * BNC) return;
  int r = (int)(gid / BNC), col = (int)(gid % BNC);
  float v;
  if (r < 512) { int b = col / NPTS; v = fg2[b * 512 + r]; }
  else v = x2[(size_t)(r - 512) * BNC + col];
  feat[gid] = v;
}

// row softmax over N; one block per row
__global__ void k_softmax_row(float* __restrict__ M)
{
  float* row = M + (size_t)blockIdx.x * NPTS;
  __shared__ float sh[256];
  float mx = -1e30f;
  for (int i = threadIdx.x; i < NPTS; i += 256) mx = fmaxf(mx, row[i]);
  sh[threadIdx.x] = mx; __syncthreads();
  for (int o = 128; o > 0; o >>= 1) {
    if (threadIdx.x < o) sh[threadIdx.x] = fmaxf(sh[threadIdx.x], sh[threadIdx.x + o]);
    __syncthreads();
  }
  mx = sh[0]; __syncthreads();
  float s = 0.f;
  for (int i = threadIdx.x; i < NPTS; i += 256) { float e = expf(row[i] - mx); row[i] = e; s += e; }
  sh[threadIdx.x] = s; __syncthreads();
  for (int o = 128; o > 0; o >>= 1) {
    if (threadIdx.x < o) sh[threadIdx.x] += sh[threadIdx.x + o];
    __syncthreads();
  }
  float inv = 1.f / sh[0];
  for (int i = threadIdx.x; i < NPTS; i += 256) row[i] *= inv;
}

__global__ void k_residual(float* __restrict__ feat, const float* __restrict__ o2,
                           const float* __restrict__ gamma, long n)
{
  long i = (long)blockIdx.x * 256 + threadIdx.x;
  if (i < n) feat[i] += gamma[0] * o2[i];
}

// all 14 output heads fused; weights staged via TDM (async tensor DMA) when available
__global__ void k_heads(const float* __restrict__ H,
                        const float* hs, const float* hbs,
                        const float* hr, const float* hbr,
                        const float* ho, const float* hbo,
                        const float* hsh, const float* hbsh,
                        const float* hx, const float* hbx,
                        float* __restrict__ out)
{
  __shared__ float wsh[14 * 128];
  __shared__ float bsh[14];
  int tid = threadIdx.x;
#if USE_TDM
  if (tid < 32) {   // wave 0 drives the Tensor Data Mover
    unsigned base = (unsigned)(size_t)&wsh[0];
    tdm_load_1d(base,                 hs,  3 * 128);
    tdm_load_1d(base + 3 * 128 * 4,   hr,  4 * 128);
    tdm_load_1d(base + 7 * 128 * 4,   ho,  1 * 128);
    tdm_load_1d(base + 8 * 128 * 4,   hsh, 3 * 128);
    tdm_load_1d(base + 11 * 128 * 4,  hx,  3 * 128);
    __builtin_amdgcn_s_wait_tensorcnt(0);
  }
#else
  for (int i = tid; i < 3 * 128; i += 256) wsh[0 * 128 + i]  = hs[i];
  for (int i = tid; i < 4 * 128; i += 256) wsh[3 * 128 + i]  = hr[i];
  for (int i = tid; i < 1 * 128; i += 256) wsh[7 * 128 + i]  = ho[i];
  for (int i = tid; i < 3 * 128; i += 256) wsh[8 * 128 + i]  = hsh[i];
  for (int i = tid; i < 3 * 128; i += 256) wsh[11 * 128 + i] = hx[i];
#endif
  if (tid < 3)       bsh[tid] = hbs[tid];
  else if (tid < 7)  bsh[tid] = hbr[tid - 3];
  else if (tid < 8)  bsh[tid] = hbo[tid - 7];
  else if (tid < 11) bsh[tid] = hbsh[tid - 8];
  else if (tid < 14) bsh[tid] = hbx[tid - 11];
  __syncthreads();
  int col = blockIdx.x * 256 + tid;
  if (col >= BNC) return;
  float acc[14];
  #pragma unroll
  for (int j = 0; j < 14; ++j) acc[j] = bsh[j];
  for (int c = 0; c < 128; ++c) {
    float v = H[(size_t)c * BNC + col];
    #pragma unroll
    for (int j = 0; j < 14; ++j) acc[j] += v * wsh[j * 128 + c];
  }
  float* o = out + (size_t)col * 14;
  for (int j = 0; j < 3; ++j) o[j] = fminf(expf(acc[j]), 0.02f);                 // scaling
  float rn = sqrtf(acc[3]*acc[3] + acc[4]*acc[4] + acc[5]*acc[5] + acc[6]*acc[6]);
  rn = fmaxf(rn, 1e-12f);
  for (int j = 0; j < 4; ++j) o[3 + j] = acc[3 + j] / rn;                        // rotation
  o[7] = 1.f / (1.f + expf(-acc[7]));                                            // opacity
  for (int j = 0; j < 3; ++j) o[8 + j]  = 1.f / (1.f + expf(-acc[8 + j]));       // shs
  for (int j = 0; j < 3; ++j) o[11 + j] = tanhf(acc[11 + j] * 0.1f) * 0.6f;      // xyz
}

// ======================================================================
// Host orchestration
// ======================================================================
static inline unsigned cdiv(long a, long b) { return (unsigned)((a + b - 1) / b); }

// EdgeBlock: fin [128][BNC] -> out [128][BNC] (with fused lrelu 0.2)
static void run_edge(hipStream_t st, void* const* d_in, int pb,
                     const float* fin, float* out,
                     float* Gbuf, float* s2b, int* idxb,
                     float* ef, float* wA, float* wB, float* hb,
                     float* meanb, float* rstdb)
{
  const float* w1   = (const float*)d_in[pb + 0];
  const float* wb1  = (const float*)d_in[pb + 1];
  const float* bn1g = (const float*)d_in[pb + 2];
  const float* bn1b = (const float*)d_in[pb + 3];
  const float* w2   = (const float*)d_in[pb + 4];
  const float* wb2  = (const float*)d_in[pb + 5];
  const float* bn2g = (const float*)d_in[pb + 6];
  const float* bn2b = (const float*)d_in[pb + 7];
  const float* x1w  = (const float*)d_in[pb + 8];
  const float* xb1  = (const float*)d_in[pb + 9];
  const float* bn3g = (const float*)d_in[pb + 10];
  const float* bn3b = (const float*)d_in[pb + 11];
  const float* ow   = (const float*)d_in[pb + 12];
  const float* ob   = (const float*)d_in[pb + 13];

  // kNN: per-batch Gram via WMMA, then top-16
  k_sumsq<<<cdiv(BNC, 256), 256, 0, st>>>(fin, s2b);
  gemm<1, 0, 0>(st, fin, fin, nullptr, Gbuf, NPTS, NPTS, DIMC,
                BNC, BNC, NPTS, NPTS, NPTS, (long)NPTS * NPTS, BB);
  k_topk<<<cdiv(BB * NPTS, 256), 256, 0, st>>>(Gbuf, s2b, idxb);
  k_gather<<<cdiv((long)KNB * BNC, 256), 256, 0, st>>>(fin, idxb, ef);

  const float* diff = ef + (size_t)DIMC * KNB * BNC;   // rows 128..255

  // w = lrelu(bn(conv(diff, w1)))      [64][16][BNC]
  gemm<0, 0, 0>(st, w1, diff, wb1, wA, 64, NKBN, 128, 128, NKBN, NKBN);
  k_stats<<<64, 256, 0, st>>>(wA, meanb, rstdb, NKBN);
  k_bn_act<<<dim3(cdiv(NKBN, 256), 64), 256, 0, st>>>(wA, meanb, rstdb, bn1g, bn1b, NKBN, 0.01f);

  // w = lrelu(bn(conv(w, w2)))         [128][16][BNC]
  gemm<0, 0, 0>(st, w2, wA, wb2, wB, 128, NKBN, 64, 64, NKBN, NKBN);
  k_stats<<<128, 256, 0, st>>>(wB, meanb, rstdb, NKBN);
  k_bn_act<<<dim3(cdiv(NKBN, 256), 128), 256, 0, st>>>(wB, meanb, rstdb, bn2g, bn2b, NKBN, 0.01f);
  k_softmax_k<<<cdiv((long)DIMC * BNC, 256), 256, 0, st>>>(wB);

  // h = lrelu(bn(conv(ef, x1)))        [128][16][BNC]
  gemm<0, 0, 0>(st, x1w, ef, xb1, hb, 128, NKBN, 256, 256, NKBN, NKBN);
  k_stats<<<128, 256, 0, st>>>(hb, meanb, rstdb, NKBN);
  k_bn_act<<<dim3(cdiv(NKBN, 256), 128), 256, 0, st>>>(hb, meanb, rstdb, bn3g, bn3b, NKBN, 0.01f);

  // h *= w ; out = lrelu_0.2(ow · h + ob)
  k_mul<<<cdiv((long)128 * NKBN, 256), 256, 0, st>>>(hb, wB, (long)128 * NKBN);
  gemm<0, 0, 2>(st, ow, hb, ob, out, 128, BNC, 2048, 2048, BNC, BNC);
}

extern "C" void kernel_launch(void* const* d_in, const int* in_sizes, int n_in,
                              void* d_out, int out_size, void* d_ws, size_t ws_size,
                              hipStream_t stream)
{
  (void)in_sizes; (void)n_in; (void)out_size; (void)ws_size;
  hipStream_t st = stream;
  const float* x = (const float*)d_in[0];
  const float* z = (const float*)d_in[1];
  // param indices (setup_inputs insertion order, depth-first)
  enum { P_HW1 = 2, P_HB1, P_HW2, P_HB2, P_PCW1, P_PCB1, P_PCW2, P_PCB2,
         P_EC1 = 10, P_EC2 = 24,
         P_AD1W = 38, P_AD1B, P_AD2W, P_AD2B,
         P_GW1 = 42, P_GB1, P_GBN1G, P_GBN1B, P_GW2, P_GB2, P_GBN2G, P_GBN2B,
         P_ATTH = 50, P_ATPH, P_ATG, P_ATO, P_ATGAMMA,
         P_M1 = 55, P_MB1, P_M2, P_MB2, P_M3, P_MB3,
         P_HS = 61, P_HBS, P_HR, P_HBR, P_HO, P_HBO, P_HSH, P_HBSH, P_HX, P_HBX };
  auto P = [&](int i) { return (const float*)d_in[i]; };

  float* W = (float*)d_ws;
  size_t off = 0;
  auto alloc = [&](size_t n) { float* p = W + off; off += n; return p; };

  float* S_in   = alloc((size_t)131 * BNC);
  float* tmp128 = alloc((size_t)128 * BNC);     // style1 / pc1 scratch
  float* styleb = alloc((size_t)128 * BNC);
  float* f0     = alloc((size_t)128 * BNC);
  float* x1b    = alloc((size_t)128 * BNC);
  float* x2b    = alloc((size_t)128 * BNC);
  float* sbuf   = alloc((size_t)256 * BNC);
  float* eout   = alloc((size_t)128 * BNC);
  float* Gbuf   = alloc((size_t)BB * NPTS * NPTS);
  float* s2b    = alloc((size_t)BNC);
  int*   idxb   = (int*)alloc((size_t)16 * BNC);
  float* ef     = alloc((size_t)256 * NKBN);
  float* wA     = alloc((size_t)64 * NKBN);
  float* wB     = alloc((size_t)128 * NKBN);
  float* hb     = alloc((size_t)128 * NKBN);
  float* meanb  = alloc(1024);
  float* rstdb  = alloc(1024);
  float* fg     = alloc((size_t)BB * 128);
  float* fg1    = alloc((size_t)BB * 128);
  float* fg2    = alloc((size_t)BB * 512);
  float* featb  = alloc((size_t)CHC * BNC);
  float* th     = alloc((size_t)80 * BNC);
  float* ph     = alloc((size_t)80 * BNC);
  float* gb     = alloc((size_t)320 * BNC);
  float* obuf   = alloc((size_t)320 * BNC);
  float* o2     = alloc((size_t)CHC * BNC);
  float* h1     = alloc((size_t)128 * BNC);
  float* h2     = alloc((size_t)128 * BNC);

  // ---- style head ----
  k_prep_style<<<cdiv(BNC, 256), 256, 0, st>>>(x, z, S_in);
  gemm<0, 0, 1, 1>(st, P(P_HW1), S_in, P(P_HB1), tmp128, 128, BNC, 131, 131, BNC, BNC);
  gemm<0, 0, 1>(st, P(P_HW2), tmp128, P(P_HB2), styleb, 128, BNC, 128, 128, BNC, BNC);

  // ---- pc head ----
  k_pc1<<<cdiv(BNC, 256), 256, 0, st>>>(x, P(P_PCW1), P(P_PCB1), tmp128);
  gemm<0, 0, 1>(st, P(P_PCW2), tmp128, P(P_PCB2), f0, 128, BNC, 64, 64, BNC, BNC);

  // ---- edge block 1 + AdaIN ----
  run_edge(st, d_in, P_EC1, f0, eout, Gbuf, s2b, idxb, ef, wA, wB, hb, meanb, rstdb);
  gemm<0, 0, 0>(st, P(P_AD1W), styleb, P(P_AD1B), sbuf, 256, BNC, 128, 128, BNC, BNC);
  k_adain<<<BB * DIMC, 256, 0, st>>>(eout, sbuf, x1b);

  // ---- edge block 2 + AdaIN ----
  run_edge(st, d_in, P_EC2, x1b, eout, Gbuf, s2b, idxb, ef, wA, wB, hb, meanb, rstdb);
  gemm<0, 0, 0>(st, P(P_AD2W), styleb, P(P_AD2B), sbuf, 256, BNC, 128, 128, BNC, BNC);
  k_adain<<<BB * DIMC, 256, 0, st>>>(eout, sbuf, x2b);

  // ---- global feature path (tiny, B=4) ----
  k_rowmax<<<BB * DIMC, 256, 0, st>>>(x2b, fg);
  k_fc_small<<<cdiv(BB * 128, 256), 256, 0, st>>>(fg, P(P_GW1), P(P_GB1), fg1, BB, 128, 128);
  k_bn1d<<<1, 256, 0, st>>>(fg1, P(P_GBN1G), P(P_GBN1B), BB, 128, 0.01f);
  k_fc_small<<<cdiv(BB * 512, 256), 256, 0, st>>>(fg1, P(P_GW2), P(P_GB2), fg2, BB, 128, 512);
  k_bn1d<<<2, 256, 0, st>>>(fg2, P(P_GBN2G), P(P_GBN2B), BB, 512, 0.01f);

  // ---- feat = [broadcast fg2 ; x2] ----
  k_feat_build<<<cdiv((long)CHC * BNC, 256), 256, 0, st>>>(fg2, x2b, featb);

  // ---- self-attention ----
  gemm<0, 0, 0, 1>(st, P(P_ATTH), featb, nullptr, th, 80, BNC, CHC, CHC, BNC, BNC);
  gemm<0, 0, 0, 1>(st, P(P_ATPH), featb, nullptr, ph, 80, BNC, CHC, CHC, BNC, BNC);
  gemm<0, 0, 0>(st, P(P_ATG),  featb, nullptr, gb, 320, BNC, CHC, CHC, BNC, BNC);
  // logits[b] = th[b]^T * ph[b]  (Gbuf reused as beta)
  gemm<1, 0, 0, 1>(st, th, ph, nullptr, Gbuf, NPTS, NPTS, 80,
                   BNC, BNC, NPTS, NPTS, NPTS, (long)NPTS * NPTS, BB);
  k_softmax_row<<<BB * NPTS, 256, 0, st>>>(Gbuf);
  // o[b] = g[b] * beta[b]^T
  gemm<0, 1, 0>(st, gb, Gbuf, nullptr, obuf, 320, NPTS, NPTS,
                BNC, NPTS, BNC, NPTS, (long)NPTS * NPTS, NPTS, BB);
  gemm<0, 0, 0>(st, P(P_ATO), obuf, nullptr, o2, CHC, BNC, 320, 320, BNC, BNC);
  k_residual<<<cdiv((long)CHC * BNC, 256), 256, 0, st>>>(featb, o2, P(P_ATGAMMA), (long)CHC * BNC);

  // ---- MLP (SiLU fused) ----
  gemm<0, 0, 3>(st, P(P_M1), featb, P(P_MB1), h1, 128, BNC, CHC, CHC, BNC, BNC);
  gemm<0, 0, 3>(st, P(P_M2), h1, P(P_MB2), h2, 128, BNC, 128, 128, BNC, BNC);
  gemm<0, 0, 3>(st, P(P_M3), h2, P(P_MB3), h1, 128, BNC, 128, 128, BNC, BNC);

  // ---- fused output heads -> d_out [B,N,14] ----
  k_heads<<<cdiv(BNC, 256), 256, 0, st>>>(h1,
      P(P_HS), P(P_HBS), P(P_HR), P(P_HBR), P(P_HO), P(P_HBO),
      P(P_HSH), P(P_HBSH), P(P_HX), P(P_HBX), (float*)d_out);
}
